// VectorQuantizedVAE_52080773432119
// MI455X (gfx1250) — matrix operand order repaired and checked by reading, not compiled, not run on hardware
//
#include <hip/hip_runtime.h>

// ---------------------------------------------------------------------------
// Types for CDNA5 WMMA
// ---------------------------------------------------------------------------
typedef __bf16 bf16_t;
typedef bf16_t v16bf __attribute__((ext_vector_type(16)));
typedef float  v8f   __attribute__((ext_vector_type(8)));

#define MB  128   // M tile per block
#define NB  128   // N tile per block
#define LDT 48    // LDS tile row stride (bf16 elems); 96B = multiple of 16B

enum { MODE_3X3 = 0, MODE_K4S2 = 1, MODE_DECT = 2 };

struct ConvParams {
  const bf16_t* A;      // activation NHWC bf16
  const bf16_t* W;      // weight [Cout][K] bf16 (per-parity stacked for DECT)
  const float*  bias;   // f32 [Cout]
  const float*  res;    // residual f32 NHWC (or null)
  float*        outF;   // f32 NHWC out (or null)
  bf16_t*       outB;   // bf16 NHWC out (or null)
  int Hin, Win, Cin, cinShift;
  int Hout, Wout, Cout;
  int K, mode;
};

__device__ inline v8f zero8() {
  v8f z = {0.f, 0.f, 0.f, 0.f, 0.f, 0.f, 0.f, 0.f};
  return z;
}

union FragU { uint4 u[2]; v16bf v; };

// CDNA5 async DMA: memory -> LDS, 16 bytes per lane, tracked by ASYNCcnt.
__device__ inline void asyncLds16(void* lds, const void* g) {
  unsigned l = (unsigned)(uintptr_t)lds;                 // LDS offset = addr[31:0]
  unsigned long long ga = (unsigned long long)(uintptr_t)g;
  asm volatile("global_load_async_to_lds_b128 %0, %1, off"
               :: "v"(l), "v"(ga) : "memory");
}
__device__ inline void waitAsync0() {
  asm volatile("s_wait_asynccnt 0x0" ::: "memory");
}

__device__ inline float blockReduceSum256(float v, float* red) {
  int t = threadIdx.x;
  red[t] = v; __syncthreads();
#pragma unroll
  for (int s = 128; s > 0; s >>= 1) {
    if (t < s) red[t] += red[t + s];
    __syncthreads();
  }
  float r = red[0];
  __syncthreads();
  return r;
}

// ---------------------------------------------------------------------------
// Generic im2col conv as WMMA GEMM with double-buffered async-LDS pipeline.
// M rows enumerate (img, y, x) on a 32x32 grid (32768 rows).  256 threads =
// 8 waves; each wave owns a 32x64 sub-tile (2x4 WMMA accumulators).
// ---------------------------------------------------------------------------
__global__ __launch_bounds__(256) void conv_gemm_wmma(ConvParams p) {
  __shared__ alignas(16) bf16_t aT[2][MB * LDT];
  __shared__ alignas(16) bf16_t bT[2][NB * LDT];

  const int tid  = threadIdx.x;
  const int mblk = blockIdx.x, nblk = blockIdx.y, par = blockIdx.z;
  const int py = par >> 1, px = par & 1;
  const bf16_t* Wp = p.W + (size_t)par * p.Cout * p.K;

  const int wave = tid >> 5, lane = tid & 31;
  const int wm = wave >> 1, wn = wave & 1;
  const int l16 = lane & 15, lh = lane >> 4;

  v8f acc[2][4];
#pragma unroll
  for (int a = 0; a < 2; a++)
#pragma unroll
    for (int b = 0; b < 4; b++) acc[a][b] = zero8();

  // loader assignment (constant over k loop)
  const int r = tid >> 1, half = tid & 1;
  const int rg  = mblk * MB + r;
  const int img = rg >> 10, yy = (rg >> 5) & 31, xx = rg & 31;
  const int nrow = nblk * NB + r;
  const int nk = p.K >> 5;

  auto issue = [&](int it) {
    const int buf = it & 1;
    const int kk = (it << 5) + half * 16;
    const int t = kk >> p.cinShift;
    const int cin = kk & (p.Cin - 1);
    int iy, ix;
    if (p.mode == MODE_3X3)       { iy = yy + t / 3 - 1;          ix = xx + t % 3 - 1; }
    else if (p.mode == MODE_K4S2) { iy = 2 * yy + (t >> 2) - 1;   ix = 2 * xx + (t & 3) - 1; }
    else                          { iy = yy + py + (t >> 1) - 1;  ix = xx + px + (t & 1) - 1; }
    bf16_t* aDst = &aT[buf][r * LDT + half * 16];
    if ((unsigned)iy < (unsigned)p.Hin && (unsigned)ix < (unsigned)p.Win) {
      const bf16_t* src = p.A + (((size_t)img * p.Hin + iy) * p.Win + ix) * p.Cin + cin;
      asyncLds16(aDst, src);
      asyncLds16(aDst + 8, src + 8);
    } else {
      uint4 z = {0u, 0u, 0u, 0u};
      uint4* ad = (uint4*)aDst; ad[0] = z; ad[1] = z;   // halo: zero-fill (compiler-visible)
    }
    const bf16_t* bs = Wp + (size_t)nrow * p.K + kk;
    bf16_t* bDst = &bT[buf][r * LDT + half * 16];
    asyncLds16(bDst, bs);
    asyncLds16(bDst + 8, bs + 8);
  };

  issue(0);
  for (int it = 0; it < nk; ++it) {
    waitAsync0();        // own async copies (current buffer) complete
    __syncthreads();     // all waves' copies complete; prior reads of other buffer done
    if (it + 1 < nk) issue(it + 1);   // overlap next tile DMA with compute
    const int buf = it & 1;

    v16bf aF[2], bF[4];
#pragma unroll
    for (int fm = 0; fm < 2; fm++) {
      const int m = wm * 32 + fm * 16 + l16;
      const uint4* pa = (const uint4*)(&aT[buf][m * LDT + lh * 8]);
      FragU f; f.u[0] = pa[0]; f.u[1] = pa[2];   // k = lh*8 + [0..7], 16 + lh*8 + [0..7]
      aF[fm] = f.v;
    }
#pragma unroll
    for (int fn = 0; fn < 4; fn++) {
      const int n = wn * 64 + fn * 16 + l16;
      const uint4* pb = (const uint4*)(&bT[buf][n * LDT + lh * 8]);
      FragU f; f.u[0] = pb[0]; f.u[1] = pb[2];   // k = lh*8 + [0..7], 16 + lh*8 + [0..7]
      bF[fn] = f.v;
    }
#pragma unroll
    for (int fm = 0; fm < 2; fm++)
#pragma unroll
      for (int fn = 0; fn < 4; fn++)
        acc[fm][fn] = __builtin_amdgcn_wmma_f32_16x16x32_bf16(
            false, aF[fm], false, bF[fn], (short)0, acc[fm][fn], false, false);
  }

  // epilogue: bias + optional residual + ReLU, scattered NHWC stores
#pragma unroll
  for (int fm = 0; fm < 2; fm++) {
#pragma unroll
    for (int fn = 0; fn < 4; fn++) {
      const int n = nblk * NB + wn * 64 + fn * 16 + l16;
      const float bv = p.bias[n];
#pragma unroll
      for (int j = 0; j < 8; j++) {
        const int ml  = wm * 32 + fm * 16 + lh * 8 + j;
        const int rg2 = mblk * MB + ml;
        const int im = rg2 >> 10, y2 = (rg2 >> 5) & 31, x2 = rg2 & 31;
        int oy, ox;
        if (p.mode == MODE_DECT) { oy = 2 * y2 + py; ox = 2 * x2 + px; }
        else                     { oy = y2;          ox = x2; }
        const size_t off = (((size_t)im * p.Hout + oy) * p.Wout + ox) * p.Cout + n;
        float v = acc[fm][fn][j] + bv;
        if (p.res)  v += p.res[off];
        v = fmaxf(v, 0.f);
        if (p.outF) p.outF[off] = v;
        if (p.outB) p.outB[off] = (bf16_t)v;
      }
    }
  }
}

// ---------------------------------------------------------------------------
// Codebook nearest-neighbor: WMMA GEMM e . c^T fused with argmin over
// score = ||c||^2 - 2 e.c   (row term is argmin-invariant).
// ---------------------------------------------------------------------------
__global__ __launch_bounds__(256) void vq_argmin_kernel(const bf16_t* __restrict__ ze,
                                                        const bf16_t* __restrict__ cb,
                                                        const float* __restrict__ c2,
                                                        int* __restrict__ idxI,
                                                        float* __restrict__ idxF) {
  __shared__ alignas(16) bf16_t aT[2][128 * LDT];
  __shared__ alignas(16) bf16_t bT[2][128 * LDT];
  __shared__ float candV[128 * 2];
  __shared__ int   candI[128 * 2];

  const int tid = threadIdx.x;
  const int wave = tid >> 5, lane = tid & 31;
  const int wm = wave >> 1, wn = wave & 1;
  const int l16 = lane & 15, lh = lane >> 4;
  const int r = tid >> 1, half = tid & 1;
  const size_t rowA = (size_t)blockIdx.x * 128 + r;

  float bestv = 3.4e38f; int besti = 0;

  auto issue = [&](int nc, int it) {
    const int buf = it & 1;
    const int kk = (it << 5) + half * 16;
    const bf16_t* as = ze + rowA * 256 + kk;
    bf16_t* ad = &aT[buf][r * LDT + half * 16];
    asyncLds16(ad, as);
    asyncLds16(ad + 8, as + 8);
    const bf16_t* bs = cb + ((size_t)(nc * 128 + r)) * 256 + kk;
    bf16_t* bd = &bT[buf][r * LDT + half * 16];
    asyncLds16(bd, bs);
    asyncLds16(bd + 8, bs + 8);
  };

  for (int nc = 0; nc < 8; nc++) {
    v8f acc[2][4];
#pragma unroll
    for (int a = 0; a < 2; a++)
#pragma unroll
      for (int b = 0; b < 4; b++) acc[a][b] = zero8();

    issue(nc, 0);
    for (int it = 0; it < 8; ++it) {
      waitAsync0();
      __syncthreads();
      if (it + 1 < 8) issue(nc, it + 1);
      const int buf = it & 1;

      v16bf aF[2], bF[4];
#pragma unroll
      for (int fm = 0; fm < 2; fm++) {
        const int m = wm * 32 + fm * 16 + l16;
        const uint4* pa = (const uint4*)(&aT[buf][m * LDT + lh * 8]);
        FragU f; f.u[0] = pa[0]; f.u[1] = pa[2];
        aF[fm] = f.v;
      }
#pragma unroll
      for (int fn = 0; fn < 4; fn++) {
        const int n = wn * 64 + fn * 16 + l16;
        const uint4* pb = (const uint4*)(&bT[buf][n * LDT + lh * 8]);
        FragU f; f.u[0] = pb[0]; f.u[1] = pb[2];
        bF[fn] = f.v;
      }
#pragma unroll
      for (int fm = 0; fm < 2; fm++)
#pragma unroll
        for (int fn = 0; fn < 4; fn++)
          acc[fm][fn] = __builtin_amdgcn_wmma_f32_16x16x32_bf16(
              false, aF[fm], false, bF[fn], (short)0, acc[fm][fn], false, false);
    }

    // per-row argmin over this 128-col block: in-wave shuffle reduce (16-lane groups)
#pragma unroll
    for (int fm = 0; fm < 2; fm++) {
#pragma unroll
      for (int j = 0; j < 8; j++) {
        float s = 3.4e38f; int ci = 0;
#pragma unroll
        for (int fn = 0; fn < 4; fn++) {
          const int col = nc * 128 + wn * 64 + fn * 16 + l16;
          const float d = c2[col] - 2.f * acc[fm][fn][j];
          if (d < s) { s = d; ci = col; }
        }
#pragma unroll
        for (int msk = 8; msk >= 1; msk >>= 1) {
          const float os = __shfl_xor(s, msk, 32);
          const int   oi = __shfl_xor(ci, msk, 32);
          if (os < s || (os == s && oi < ci)) { s = os; ci = oi; }
        }
        if (l16 == 0) {
          const int m = wm * 32 + fm * 16 + lh * 8 + j;
          candV[m * 2 + wn] = s; candI[m * 2 + wn] = ci;
        }
      }
    }
    __syncthreads();
    if (tid < 128) {
#pragma unroll
      for (int w2 = 0; w2 < 2; w2++) {
        const float cv = candV[tid * 2 + w2];
        const int  cix = candI[tid * 2 + w2];
        if (cv < bestv || (cv == bestv && cix < besti)) { bestv = cv; besti = cix; }
      }
    }
  }

  if (tid < 128) {
    const size_t row = (size_t)blockIdx.x * 128 + tid;
    idxI[row] = besti;
    idxF[row] = (float)besti;
  }
}

// ---------------------------------------------------------------------------
// Householder rotation + VQ loss + EMA accumulators.  One block per row.
// ---------------------------------------------------------------------------
__global__ __launch_bounds__(256) void householder_kernel(const float* __restrict__ zeF,
                                                          const float* __restrict__ cbF,
                                                          const int* __restrict__ idxI,
                                                          float* __restrict__ ztF,
                                                          bf16_t* __restrict__ ztB,
                                                          float* counts, float* dwacc, float* lossAcc) {
  __shared__ float red[256];
  const int rrow = blockIdx.x, c = threadIdx.x;
  const int idx = idxI[rrow];
  const float e = zeF[(size_t)rrow * 256 + c];
  const float q = cbF[(size_t)idx * 256 + c];
  const float ee = blockReduceSum256(e * e, red);
  const float qq = blockReduceSum256(q * q, red);
  const float en = e / fmaxf(sqrtf(ee), 1e-12f);
  const float qn = q / fmaxf(sqrtf(qq), 1e-12f);
  float v = en - qn;
  const float vv = blockReduceSum256(v * v, red);
  const float vn = sqrtf(vv);
  if (vn > 1e-5f) v = v / (vn + 1e-5f);
  const float ve = blockReduceSum256(v * e, red);
  const float zt = e - 2.f * v * ve;
  ztF[(size_t)rrow * 256 + c] = zt;
  ztB[(size_t)rrow * 256 + c] = (bf16_t)zt;
  const float d = e - q;
  const float ls = blockReduceSum256(d * d, red);
  if (c == 0) { atomicAdd(lossAcc + 1, ls); atomicAdd(counts + idx, 1.0f); }
  atomicAdd(dwacc + (size_t)idx * 256 + c, e);
}

// ---------------------------------------------------------------------------
// Small direct convs (tiny K: not worth WMMA)
// ---------------------------------------------------------------------------
__global__ __launch_bounds__(256) void enc_conv1_kernel(const float* __restrict__ X,
                                                        const float* __restrict__ w,
                                                        const float* __restrict__ b,
                                                        bf16_t* __restrict__ out) {
  const int gid = blockIdx.x * 256 + threadIdx.x;   // 32*64*64*128
  const int co = gid & 127;
  const int x  = (gid >> 7) & 63;
  const int y  = (gid >> 13) & 63;
  const int n  = gid >> 19;
  float acc = b[co];
#pragma unroll
  for (int ky = 0; ky < 4; ky++) {
    const int iy = 2 * y + ky - 1;
    if ((unsigned)iy >= 128u) continue;
#pragma unroll
    for (int kx = 0; kx < 4; kx++) {
      const int ix = 2 * x + kx - 1;
      if ((unsigned)ix >= 128u) continue;
#pragma unroll
      for (int ci = 0; ci < 3; ci++)
        acc += X[(((size_t)n * 3 + ci) * 128 + iy) * 128 + ix] * w[((co * 3 + ci) * 4 + ky) * 4 + kx];
    }
  }
  out[gid] = (bf16_t)fmaxf(acc, 0.f);
}

__global__ __launch_bounds__(256) void dec_final_kernel(const float* __restrict__ dd,
                                                        const float* __restrict__ w,
                                                        const float* __restrict__ b,
                                                        const float* __restrict__ X,
                                                        float* __restrict__ xrec, float* lossAcc) {
  __shared__ float red[256];
  const int gid = blockIdx.x * 256 + threadIdx.x;   // 32*3*128*128 NCHW
  int t = gid;
  const int ox = t & 127; t >>= 7;
  const int oy = t & 127; t >>= 7;
  const int co = t % 3;
  const int n  = t / 3;
  float acc = b[co];
  const int py = oy & 1, px = ox & 1;
#pragma unroll
  for (int jy = 0; jy < 2; jy++) {
    const int ky = py + 2 * jy;
    const int iy = (oy + ky - 2) >> 1;
    if ((unsigned)iy >= 64u) continue;
#pragma unroll
    for (int jx = 0; jx < 2; jx++) {
      const int kx = px + 2 * jx;
      const int ix = (ox + kx - 2) >> 1;
      if ((unsigned)ix >= 64u) continue;
      const float* ip = dd + (((size_t)n * 64 + iy) * 64 + ix) * 128;
      const float* wp = w + (size_t)co * 2048 + ky * 4 + kx;
      float s = 0.f;
      for (int ci = 0; ci < 128; ci++) s += ip[ci] * wp[ci * 16];
      acc += s;
    }
  }
  const float xr = tanhf(acc);
  xrec[gid] = xr;
  const float df = xr - X[gid];
  const float ls = blockReduceSum256(df * df, red);
  if (threadIdx.x == 0) atomicAdd(lossAcc, ls);
}

// ---------------------------------------------------------------------------
// Weight transforms f32 OIHW -> bf16 [Cout][K] matching the A-gather order
// ---------------------------------------------------------------------------
__global__ void wt_res_kernel(const float* __restrict__ w, bf16_t* __restrict__ out) {
  const int gid = blockIdx.x * 256 + threadIdx.x;   // 256*2304
  const int co = gid / 2304;
  const int rem = gid - co * 2304;
  const int t = rem >> 8, cin = rem & 255;
  const int dy = t / 3, dx = t - dy * 3;
  out[gid] = (bf16_t)w[((co * 256 + cin) * 3 + dy) * 3 + dx];
}
__global__ void wt_enc2_kernel(const float* __restrict__ w, bf16_t* __restrict__ out) {
  const int gid = blockIdx.x * 256 + threadIdx.x;   // 256*2048
  const int co = gid >> 11;
  const int rem = gid & 2047;
  const int t = rem >> 7, cin = rem & 127;
  const int dy = t >> 2, dx = t & 3;
  out[gid] = (bf16_t)w[((co * 128 + cin) * 4 + dy) * 4 + dx];
}
__global__ void wt_dect_kernel(const float* __restrict__ w, bf16_t* __restrict__ out) {
  const int gid = blockIdx.x * 256 + threadIdx.x;   // 4*128*1024
  const int p   = gid >> 17;
  const int rem = gid & 131071;
  const int co  = rem >> 10;
  const int r2  = rem & 1023;
  const int t = r2 >> 8, cin = r2 & 255;
  const int py = p >> 1, px = p & 1;
  const int ky = py + 2 * (t >> 1), kx = px + 2 * (t & 1);
  out[gid] = (bf16_t)w[((co * 256 + cin) * 4 + ky) * 4 + kx];
}
__global__ void cvt_cb_kernel(const float* __restrict__ cb, bf16_t* __restrict__ out) {
  const int gid = blockIdx.x * 256 + threadIdx.x;   // 262144
  out[gid] = (bf16_t)cb[gid];
}
__global__ void c2_kernel(const float* __restrict__ cb, float* __restrict__ c2) {
  const int k = blockIdx.x * 256 + threadIdx.x;     // 1024
  float s = 0.f;
  for (int c = 0; c < 256; c++) { const float v = cb[(size_t)k * 256 + c]; s += v * v; }
  c2[k] = s;
}

// ---------------------------------------------------------------------------
// Finalize: new_ecs, n, perplexity, scalar losses; then new codebook.
// ---------------------------------------------------------------------------
__global__ __launch_bounds__(1024) void finalize_kernel(const float* counts, const float* emaCS,
                                                        const float* lossAcc,
                                                        float* o_scal, float* o_necs, float* nbuf) {
  __shared__ float red[1024];
  const int k = threadIdx.x;
  const float cnt  = counts[k];
  const float necs = 0.99f * emaCS[k] + 0.01f * cnt;
  o_necs[k] = necs;
  red[k] = necs; __syncthreads();
  for (int s = 512; s > 0; s >>= 1) { if (k < s) red[k] += red[k + s]; __syncthreads(); }
  const float n = red[0]; __syncthreads();
  const float pp = cnt / 32768.f;
  red[k] = pp * logf(pp + 1e-10f); __syncthreads();
  for (int s = 512; s > 0; s >>= 1) { if (k < s) red[k] += red[k + s]; __syncthreads(); }
  const float plsum = red[0];
  if (k == 0) {
    nbuf[0] = n;
    const float recon = lossAcc[0] / 1572864.f;
    const float vq    = lossAcc[1] / 8388608.f;
    o_scal[0] = recon + vq + 0.25f * vq;   // total
    o_scal[1] = recon;                     // recon
    o_scal[2] = vq;                        // codebook
    o_scal[3] = vq;                        // commit
    o_scal[4] = expf(-plsum);              // perplexity
  }
}

__global__ void new_codebook_kernel(const float* __restrict__ emaDW, const float* __restrict__ dwacc,
                                    const float* __restrict__ o_necs, const float* __restrict__ nbuf,
                                    float* __restrict__ o_ncb, float* __restrict__ o_nedw) {
  const int i = blockIdx.x * 256 + threadIdx.x;   // 262144
  const int k = i >> 8;
  const float dwv = 0.99f * emaDW[i] + 0.01f * dwacc[i];
  o_nedw[i] = dwv;
  const float necs = o_necs[k];
  const float n = nbuf[0];
  const float cs = (necs + 1e-5f) / (n + 1024.f * 1e-5f) * n;
  o_ncb[i] = dwv / cs;
}

// ---------------------------------------------------------------------------
// Host orchestration
// ---------------------------------------------------------------------------
extern "C" void kernel_launch(void* const* d_in, const int* in_sizes, int n_in,
                              void* d_out, int out_size, void* d_ws, size_t ws_size,
                              hipStream_t stream) {
  (void)in_sizes; (void)n_in; (void)out_size; (void)ws_size;

  const float* X      = (const float*)d_in[0];
  const float* enc_w1 = (const float*)d_in[1];
  const float* enc_b1 = (const float*)d_in[2];
  const float* enc_w2 = (const float*)d_in[3];
  const float* enc_b2 = (const float*)d_in[4];
  const float* rw[8]; const float* rb[8];
  for (int i = 0; i < 4; i++) {
    rw[2 * i]     = (const float*)d_in[5 + 4 * i];
    rb[2 * i]     = (const float*)d_in[6 + 4 * i];
    rw[2 * i + 1] = (const float*)d_in[7 + 4 * i];
    rb[2 * i + 1] = (const float*)d_in[8 + 4 * i];
  }
  const float* dec_w1   = (const float*)d_in[21];
  const float* dec_b1   = (const float*)d_in[22];
  const float* dec_w2   = (const float*)d_in[23];
  const float* dec_b2   = (const float*)d_in[24];
  const float* codebook = (const float*)d_in[25];
  const float* emaCS    = (const float*)d_in[26];
  const float* emaDW    = (const float*)d_in[27];

  float* out = (float*)d_out;
  float* o_xrec = out;                 // 1572864
  float* o_scal = out + 1572864;       // 5 scalars
  float* o_idx  = out + 1572869;       // 32768
  float* o_ncb  = out + 1605637;       // 262144
  float* o_necs = out + 1867781;       // 1024
  float* o_nedw = out + 1868805;       // 262144

  char* ws = (char*)d_ws; size_t off = 0;
  auto alloc = [&](size_t bytes) -> void* {
    off = (off + 255) & ~(size_t)255;
    void* p = ws + off; off += bytes; return p;
  };

  const size_t ACT = 32ull * 32 * 32 * 256;           // 8388608 elems
  bf16_t* h1b   = (bf16_t*)alloc(32ull * 64 * 64 * 128 * 2);
  float*  actF0 = (float*) alloc(ACT * 4);
  float*  actF1 = (float*) alloc(ACT * 4);
  bf16_t* actB0 = (bf16_t*)alloc(ACT * 2);
  bf16_t* actB1 = (bf16_t*)alloc(ACT * 2);
  bf16_t* midB  = (bf16_t*)alloc(ACT * 2);
  float*  zeF   = (float*) alloc(ACT * 4);
  bf16_t* zeB   = (bf16_t*)alloc(ACT * 2);
  float*  ztF   = (float*) alloc(ACT * 4);
  bf16_t* ztB   = (bf16_t*)alloc(ACT * 2);
  float*  ddF   = (float*) alloc(32ull * 64 * 64 * 128 * 4);
  bf16_t* wtEnc2 = (bf16_t*)alloc(256ull * 2048 * 2);
  bf16_t* wtRes  = (bf16_t*)alloc(8ull * 256 * 2304 * 2);
  bf16_t* wtDect = (bf16_t*)alloc(4ull * 128 * 1024 * 2);
  bf16_t* cbB    = (bf16_t*)alloc(1024ull * 256 * 2);
  float*  c2     = (float*) alloc(1024 * 4);
  int*    idxI   = (int*)   alloc(32768 * 4);
  float*  zreg   = (float*) alloc((1024 + 262144 + 8) * 4);
  float* counts  = zreg;
  float* dwacc   = zreg + 1024;
  float* lossAcc = zreg + 1024 + 262144;
  float* nbuf    = lossAcc + 4;

  hipMemsetAsync(zreg, 0, (1024 + 262144 + 8) * 4, stream);

  // weight transforms
  wt_enc2_kernel<<<2048, 256, 0, stream>>>(enc_w2, wtEnc2);
  for (int j = 0; j < 8; j++)
    wt_res_kernel<<<2304, 256, 0, stream>>>(rw[j], wtRes + (size_t)j * 589824);
  wt_dect_kernel<<<2048, 256, 0, stream>>>(dec_w1, wtDect);
  cvt_cb_kernel<<<1024, 256, 0, stream>>>(codebook, cbB);
  c2_kernel<<<4, 256, 0, stream>>>(codebook, c2);

  // encoder conv1 (direct, tiny K)
  enc_conv1_kernel<<<65536, 256, 0, stream>>>(X, enc_w1, enc_b1, h1b);

  ConvParams cp{};
  // encoder conv2: 128->256, k4 s2
  cp.A = h1b; cp.W = wtEnc2; cp.bias = enc_b2; cp.res = nullptr;
  cp.outF = actF0; cp.outB = actB0;
  cp.Hin = 64; cp.Win = 64; cp.Cin = 128; cp.cinShift = 7;
  cp.Hout = 32; cp.Wout = 32; cp.Cout = 256; cp.K = 2048; cp.mode = MODE_K4S2;
  conv_gemm_wmma<<<dim3(256, 2, 1), 256, 0, stream>>>(cp);

  auto resMid = [&](const bf16_t* Ain, int widx, const float* bias) {
    ConvParams q{};
    q.A = Ain; q.W = wtRes + (size_t)widx * 589824; q.bias = bias; q.res = nullptr;
    q.outF = nullptr; q.outB = midB;
    q.Hin = 32; q.Win = 32; q.Cin = 256; q.cinShift = 8;
    q.Hout = 32; q.Wout = 32; q.Cout = 256; q.K = 2304; q.mode = MODE_3X3;
    conv_gemm_wmma<<<dim3(256, 2, 1), 256, 0, stream>>>(q);
  };
  auto resOut = [&](int widx, const float* bias, const float* resF, float* oF, bf16_t* oB) {
    ConvParams q{};
    q.A = midB; q.W = wtRes + (size_t)widx * 589824; q.bias = bias; q.res = resF;
    q.outF = oF; q.outB = oB;
    q.Hin = 32; q.Win = 32; q.Cin = 256; q.cinShift = 8;
    q.Hout = 32; q.Wout = 32; q.Cout = 256; q.K = 2304; q.mode = MODE_3X3;
    conv_gemm_wmma<<<dim3(256, 2, 1), 256, 0, stream>>>(q);
  };

  // encoder resblocks
  resMid(actB0, 0, rb[0]); resOut(1, rb[1], actF0, actF1, actB1);   // enc_r1 -> h3
  resMid(actB1, 2, rb[2]); resOut(3, rb[3], actF1, zeF, zeB);       // enc_r2 -> z_e

  // vector quantization
  vq_argmin_kernel<<<256, 256, 0, stream>>>(zeB, cbB, c2, idxI, o_idx);
  householder_kernel<<<32768, 256, 0, stream>>>(zeF, codebook, idxI, ztF, ztB,
                                                counts, dwacc, lossAcc);

  // decoder resblocks (input = zq_t, since z_q_st == zq_t numerically)
  resMid(ztB, 4, rb[4]); resOut(5, rb[5], ztF, actF0, actB0);       // dec_r1 -> d1
  resMid(actB0, 6, rb[6]); resOut(7, rb[7], actF0, actF1, actB1);   // dec_r2 -> d2

  // decoder conv-transpose 1 (256->128) as 4 parity-class GEMMs
  ConvParams cd{};
  cd.A = actB1; cd.W = wtDect; cd.bias = dec_b1; cd.res = nullptr;
  cd.outF = ddF; cd.outB = nullptr;
  cd.Hin = 32; cd.Win = 32; cd.Cin = 256; cd.cinShift = 8;
  cd.Hout = 64; cd.Wout = 64; cd.Cout = 128; cd.K = 1024; cd.mode = MODE_DECT;
  conv_gemm_wmma<<<dim3(256, 1, 4), 256, 0, stream>>>(cd);

  // final conv-transpose (128->3) + tanh + recon loss
  dec_final_kernel<<<6144, 256, 0, stream>>>(ddF, dec_w2, dec_b2, X, o_xrec, lossAcc);

  // losses / EMA finalize / new codebook
  finalize_kernel<<<1, 1024, 0, stream>>>(counts, emaCS, lossAcc, o_scal, o_necs, nbuf);
  new_codebook_kernel<<<1024, 256, 0, stream>>>(emaDW, dwacc, o_necs, nbuf, o_ncb, o_nedw);
}
